// RoadGNN_62577673503741
// MI455X (gfx1250) — matrix operand order repaired and checked by reading, not compiled, not run on hardware
//
#include <hip/hip_runtime.h>
#include <hip/hip_bf16.h>

typedef __attribute__((ext_vector_type(16))) _Float16 v16h;
typedef __attribute__((ext_vector_type(8)))  _Float16 v8h;
typedef __attribute__((ext_vector_type(8)))  float    v8f;

#define DFEAT 128
#define HEADS 4
#define NEG_SLOPE 0.2f

// ---- monotonic float<->uint encoding for atomicMax-based segment max ----
__device__ __forceinline__ unsigned enc_f32(float f) {
  unsigned u = __float_as_uint(f);
  return (u & 0x80000000u) ? ~u : (u | 0x80000000u);
}
__device__ __forceinline__ float dec_f32(unsigned e) {
  unsigned u = (e & 0x80000000u) ? (e & 0x7FFFFFFFu) : ~e;
  return __uint_as_float(u);
}

// ============================================================================
// Repack W[128,128] f32 (row-major, xp = h @ W) into WMMA-fragment-ordered
// f16: wf[(kt*8 + nt)*32 + lane][16 halves] = exactly the v16h that lane
// of the wave owning n-tile nt needs at k-tile kt.  32KB per layer,
// L2-resident for the whole GEMM.
// ============================================================================
__global__ void gat_repack_w(const float* __restrict__ W, _Float16* __restrict__ wf)
{
  int t = blockIdx.x * blockDim.x + threadIdx.x;      // 4*8*32 = 1024 frags
  if (t >= 4 * 8 * 32) return;
  int lane = t & 31;
  int nt   = (t >> 5) & 7;
  int kt   = t >> 8;
  int l16  = lane & 15;
  int kbB  = (lane & 16) ? 16 : 0;
  const int base = (kt * 32 + kbB) * DFEAT + nt * 16 + l16;
  _Float16* o = wf + (size_t)t * 16;
#pragma unroll
  for (int j = 0; j < 16; ++j)
    o[j] = (_Float16)W[base + j * DFEAT];
}

// ============================================================================
// GEMM: out[n,128] = A[n,128] @ W (+ bias), W pre-repacked to f16 fragments.
// Block = 16 rows, 8 waves; wave w owns the 16x16 tile at cols [16w,16w+16).
// A tile staged in LDS as f16 with K-octets in natural order {0,1,2,3} per 32-group
// so each lane's 16 A-halves are one contiguous 32B read.
// Inner loop: 2x ds_load_b128 + 2x global_load_b128 + 1x v_wmma per k-step.
// ============================================================================
__global__ void __launch_bounds__(256) gat_gemm128_wmma(
    const float* __restrict__ A, const _Float16* __restrict__ wf,
    const float* __restrict__ bias, float* __restrict__ out, int n)
{
  __shared__ __align__(32) _Float16 As[16 * DFEAT];
  const int row0 = blockIdx.x * 16;

  // ---- fill: one K-octet (8 floats -> 8 halves) per thread, coalesced ----
  {
    const int o    = threadIdx.x;        // 16 rows * 16 octets = 256
    const int row  = o >> 4;
    const int kOct = o & 15;
    const int g    = kOct >> 2;          // 32-wide K group
    const int jOct = kOct & 3;
    const int pOct = jOct;                              // {0,1,2,3}
    const int grow = row0 + row;
    v8h hv = {};
    if (grow < n) {
      const float* ap = A + (size_t)grow * DFEAT + kOct * 8;
#pragma unroll
      for (int j = 0; j < 8; ++j) hv[j] = (_Float16)ap[j];
    }
    *(v8h*)(As + row * DFEAT + g * 32 + pOct * 8) = hv;
    // warm L2 for the next 16-row tile (streamed once, never revisited)
    if (row0 + 16 < n)
      __builtin_prefetch(A + (size_t)(row0 + 16) * DFEAT + threadIdx.x * 8, 0, 1);
  }
  __syncthreads();

  const int wave = threadIdx.x >> 5;
  const int lane = threadIdx.x & 31;
  const int hi   = lane >> 4;
  const int l16  = lane & 15;

  v8f acc = {};
#pragma unroll
  for (int kt = 0; kt < 4; ++kt) {
    v16h a = *(const v16h*)(As + l16 * DFEAT + kt * 32 + hi * 16);
    v16h b = *(const v16h*)(wf + (size_t)(((kt * 8 + wave) * 32) + lane) * 16);
    acc = __builtin_amdgcn_wmma_f32_16x16x32_f16(
        false, a, false, b, (short)0, acc, false, false);
  }

  const int   col = wave * 16 + l16;
  const float bv  = bias ? bias[col] : 0.0f;
  const int   rb  = row0 + (hi ? 8 : 0);
  if (row0 + 16 <= n) {
    // full tile (always taken when n % 16 == 0): straight coalesced stores
#pragma unroll
    for (int r = 0; r < 8; ++r)
      out[(size_t)(rb + r) * DFEAT + col] = acc[r] + bv;
  } else {
    // ragged tail tile
#pragma unroll
    for (int r = 0; r < 8; ++r) {
      int row = rb + r;
      if (row < n) out[(size_t)row * DFEAT + col] = acc[r] + bv;
    }
  }
}

// ============================================================================
// Per-node attention logits: al_s[n,h] = dot(xp[n,h,:], a_src[h,:]) etc.
// ============================================================================
__global__ void gat_attn_logits(const float* __restrict__ xp,
                                const float* __restrict__ a_src,
                                const float* __restrict__ a_dst,
                                float* __restrict__ als, float* __restrict__ ald,
                                int n)
{
  int idx = blockIdx.x * blockDim.x + threadIdx.x;
  if (idx >= n * HEADS) return;
  int node = idx >> 2, h = idx & 3;
  const float* xr = xp + (size_t)node * DFEAT + h * 32;
  float ss = 0.0f, sd = 0.0f;
#pragma unroll
  for (int c = 0; c < 32; ++c) {
    float v = xr[c];
    ss += v * a_src[h * 32 + c];
    sd += v * a_dst[h * 32 + c];
  }
  als[idx] = ss;
  ald[idx] = sd;
}

// ============================================================================
// Init per-layer state: mx=enc(-inf)=0, s=0, accum[n,d]=bias[d]
// ============================================================================
__global__ void gat_init(unsigned* __restrict__ mx, float* __restrict__ s,
                         float* __restrict__ accum, const float* __restrict__ bias,
                         int n)
{
  int i = blockIdx.x * blockDim.x + threadIdx.x;
  if (i < n * HEADS) { mx[i] = 0u; s[i] = 0.0f; }
  if (i < n * DFEAT) accum[i] = bias[i & (DFEAT - 1)];
}

__device__ __forceinline__ void edge_src_dst(const int* __restrict__ ei,
                                             int e, int E, int& src, int& dst) {
  if (e < E) { src = ei[e]; dst = ei[E + e]; }
  else       { src = dst = e - E; }          // appended self loops
}

__device__ __forceinline__ float leaky(float x) {
  return (x > 0.0f) ? x : NEG_SLOPE * x;
}

// ============================================================================
// Pass 1: segment max of leaky_relu(al_s[src]+al_d[dst]) over dst
// ============================================================================
__global__ void gat_edge_max(const int* __restrict__ ei,
                             const float* __restrict__ als,
                             const float* __restrict__ ald,
                             unsigned* __restrict__ mx, int E, int Etot)
{
  int t = blockIdx.x * blockDim.x + threadIdx.x;
  if (t >= Etot * HEADS) return;
  int e = t >> 2, h = t & 3, src, dst;
  edge_src_dst(ei, e, E, src, dst);
  float lg = leaky(als[src * HEADS + h] + ald[dst * HEADS + h]);
  atomicMax(mx + dst * HEADS + h, enc_f32(lg));
}

// ============================================================================
// Pass 2: segment sum of exp(logit - max)
// ============================================================================
__global__ void gat_edge_sum(const int* __restrict__ ei,
                             const float* __restrict__ als,
                             const float* __restrict__ ald,
                             const unsigned* __restrict__ mx,
                             float* __restrict__ s, int E, int Etot)
{
  int t = blockIdx.x * blockDim.x + threadIdx.x;
  if (t >= Etot * HEADS) return;
  int e = t >> 2, h = t & 3, src, dst;
  edge_src_dst(ei, e, E, src, dst);
  int ih = dst * HEADS + h;
  float lg = leaky(als[src * HEADS + h] + ald[dst * HEADS + h]);
  atomicAdd(s + ih, __expf(lg - dec_f32(mx[ih])));
}

// ============================================================================
// Pass 3: accum[dst,:] += alpha * xp[src,:].  One wave per edge; each lane
// moves a float4 (cols lane*4 .. lane*4+3), head = lane/8.
// ============================================================================
__global__ void gat_edge_scatter(const int* __restrict__ ei,
                                 const float* __restrict__ als,
                                 const float* __restrict__ ald,
                                 const unsigned* __restrict__ mx,
                                 const float* __restrict__ s,
                                 const float* __restrict__ xp,
                                 float* __restrict__ accum, int E, int Etot)
{
  int gt   = blockIdx.x * blockDim.x + threadIdx.x;
  int e    = gt >> 5;
  int lane = gt & 31;
  if (e >= Etot) return;
  int src, dst;
  edge_src_dst(ei, e, E, src, dst);
  int h  = lane >> 3;
  int ih = dst * HEADS + h;
  float lg    = leaky(als[src * HEADS + h] + ald[dst * HEADS + h]);
  float alpha = __expf(lg - dec_f32(mx[ih])) / s[ih];
  const float4 xv = *(const float4*)(xp + (size_t)src * DFEAT + lane * 4);
  float* o = accum + (size_t)dst * DFEAT + lane * 4;
  atomicAdd(o + 0, alpha * xv.x);
  atomicAdd(o + 1, alpha * xv.y);
  atomicAdd(o + 2, alpha * xv.z);
  atomicAdd(o + 3, alpha * xv.w);
}

// ============================================================================
// Residual + ReLU (layers 1,2): acc = relu(hin + acc)
// ============================================================================
__global__ void gat_residual_relu(const float* __restrict__ hin,
                                  float* __restrict__ acc, int n)
{
  int i = blockIdx.x * blockDim.x + threadIdx.x;
  if (i >= n * DFEAT) return;
  float v = hin[i] + acc[i];
  acc[i] = v > 0.0f ? v : 0.0f;
}

// ============================================================================
extern "C" void kernel_launch(void* const* d_in, const int* in_sizes, int n_in,
                              void* d_out, int out_size, void* d_ws, size_t ws_size,
                              hipStream_t stream)
{
  const float* x   = (const float*)d_in[0];
  const int*   ei  = (const int*)d_in[1];
  const float* w0  = (const float*)d_in[2];
  const float* b0  = (const float*)d_in[3];
  const float* wL[4];  const float* asL[4];  const float* adL[4];  const float* bL[4];
  for (int i = 1; i <= 3; ++i) {
    wL[i]  = (const float*)d_in[4 * i + 0];
    asL[i] = (const float*)d_in[4 * i + 1];
    adL[i] = (const float*)d_in[4 * i + 2];
    bL[i]  = (const float*)d_in[4 * i + 3];
  }

  const int N    = in_sizes[0] / DFEAT;
  const int E    = in_sizes[1] / 2;
  const int Etot = E + N;                 // with self loops

  // ---- carve workspace ----
  char* p = (char*)d_ws;
  auto carve = [&](size_t bytes) -> char* {
    char* r = p;
    p += (bytes + 255) & ~(size_t)255;
    return r;
  };
  float*    hA  = (float*)carve((size_t)N * DFEAT * sizeof(float));
  float*    hB  = (float*)carve((size_t)N * DFEAT * sizeof(float));
  float*    xp  = (float*)carve((size_t)N * DFEAT * sizeof(float));
  float*    als = (float*)carve((size_t)N * HEADS * sizeof(float));
  float*    ald = (float*)carve((size_t)N * HEADS * sizeof(float));
  unsigned* mx  = (unsigned*)carve((size_t)N * HEADS * sizeof(unsigned));
  float*    ssm = (float*)carve((size_t)N * HEADS * sizeof(float));
  _Float16* wf[4];
  for (int i = 0; i < 4; ++i)
    wf[i] = (_Float16*)carve((size_t)4 * 8 * 32 * 16 * sizeof(_Float16)); // 32KB

  const int gRows  = (N + 15) / 16;                     // GEMM blocks (16 rows)
  const int gNode  = (N * DFEAT + 255) / 256;
  const int gNH    = (N * HEADS + 255) / 256;
  const int gEH    = (Etot * HEADS + 255) / 256;
  const int gEdgeW = (int)(((size_t)Etot * 32 + 255) / 256);

  // repack all four weight matrices to f16 fragment order
  gat_repack_w<<<4, 256, 0, stream>>>(w0, wf[0]);
  gat_repack_w<<<4, 256, 0, stream>>>(wL[1], wf[1]);
  gat_repack_w<<<4, 256, 0, stream>>>(wL[2], wf[2]);
  gat_repack_w<<<4, 256, 0, stream>>>(wL[3], wf[3]);

  // h0 = x @ w0 + b0
  gat_gemm128_wmma<<<gRows, 256, 0, stream>>>(x, wf[0], b0, hA, N);

  float* hin = hA;
  float* alt = hB;
  for (int L = 1; L <= 3; ++L) {
    float* acc = (L == 3) ? (float*)d_out : alt;

    gat_gemm128_wmma<<<gRows, 256, 0, stream>>>(hin, wf[L], nullptr, xp, N);
    gat_attn_logits<<<gNH, 256, 0, stream>>>(xp, asL[L], adL[L], als, ald, N);
    gat_init<<<gNode, 256, 0, stream>>>(mx, ssm, acc, bL[L], N);
    gat_edge_max<<<gEH, 256, 0, stream>>>(ei, als, ald, mx, E, Etot);
    gat_edge_sum<<<gEH, 256, 0, stream>>>(ei, als, ald, mx, ssm, E, Etot);
    gat_edge_scatter<<<gEdgeW, 256, 0, stream>>>(ei, als, ald, mx, ssm, xp, acc, E, Etot);

    if (L < 3) {
      gat_residual_relu<<<gNode, 256, 0, stream>>>(hin, acc, N);
      float* t = hin; hin = acc; alt = t;   // new h lives in acc; old h reusable
    }
  }
}
